// ActorCriticTransformer_53549652247150
// MI455X (gfx1250) — matrix operand order repaired and checked by reading, not compiled, not run on hardware
//
#include <hip/hip_runtime.h>

typedef __attribute__((ext_vector_type(16))) _Float16 v16h;
typedef __attribute__((ext_vector_type(8)))  float    v8f;
typedef __attribute__((ext_vector_type(4)))  unsigned v4u;
typedef __attribute__((ext_vector_type(8)))  int      v8i;
typedef __attribute__((ext_vector_type(4)))  int      v4i;

#if defined(__has_builtin)
#if __has_builtin(__builtin_amdgcn_tensor_load_to_lds) && \
    __has_builtin(__builtin_amdgcn_s_wait_tensorcnt)
#define HAVE_TDM 1
#endif
#if __has_builtin(__builtin_amdgcn_cvt_pkrtz)
#define HAVE_PKRTZ 1
#endif
#endif
#ifndef HAVE_TDM
#define HAVE_TDM 0
#endif
#ifndef HAVE_PKRTZ
#define HAVE_PKRTZ 0
#endif

namespace {
constexpr int T_ = 8, L_ = 2, B_ = 512, M_ = 128, E_ = 512, H_ = 8;
constexpr int A_ = 16, HID_ = 1024, OBS_ = 64;
constexpr int S_ = M_ + 1;      // 129 (memory + current token)
constexpr int DH_ = E_ / H_;    // 64
}

static __device__ __forceinline__ unsigned short h16(float f) {
  _Float16 h = (_Float16)f;
  return __builtin_bit_cast(unsigned short, h);
}
#if HAVE_PKRTZ
static __device__ __forceinline__ unsigned pack2(float a, float b) {
  auto p = __builtin_amdgcn_cvt_pkrtz(a, b);  // one v_cvt_pk_rtz_f16_f32
  return __builtin_bit_cast(unsigned, p);
}
#else
static __device__ __forceinline__ unsigned pack2(float a, float b) {
  return (unsigned)h16(a) | ((unsigned)h16(b) << 16);
}
#endif

union HFrag { v16h v; unsigned u[8]; };

#if HAVE_TDM
// Tensor DMA: copy a [tile_rows x 32] tile of 2-byte elements (row stride =
// strideK elements) from global into LDS, packed densely row-major.
// D# per CDNA5 ISA 8.3/8.4. global_addr is the TILE start, so tensor dims are
// the remaining extents (OOB reads -> zero-fill).
static __device__ __forceinline__ void tdm_load_w_tile(
    const unsigned short* gsrc, unsigned lds_byte_off, int strideK,
    int rem_k, int rem_rows, int tile_rows) {
  unsigned long long ga = (unsigned long long)(__UINTPTR_TYPE__)gsrc;
  v4u g0;
  g0[0] = 1u;                                             // count=1 (valid)
  g0[1] = lds_byte_off;                                   // LDS byte address
  g0[2] = (unsigned)(ga & 0xffffffffu);                   // global addr lo
  g0[3] = (unsigned)((ga >> 32) & 0x1ffffffu) | (2u << 30);  // addr hi|type=2
  v8i g1;
  g1[0] = (int)(1u << 16);                                // data_size=1 -> 2B
  g1[1] = (int)((unsigned)rem_k << 16);                   // tensor_dim0 lo16
  g1[2] = (int)((((unsigned)rem_k) >> 16) |
                ((unsigned)rem_rows << 16));              // d0 hi | d1 lo
  g1[3] = (int)((((unsigned)rem_rows) >> 16) | (32u << 16));  // d1 hi|tile_d0
  g1[4] = (int)(unsigned)tile_rows;                       // tile_dim1, d2=0
  g1[5] = (int)(unsigned)strideK;                         // dim0 stride lo32
  g1[6] = 0;
  g1[7] = 0;
  v4i gz = {0, 0, 0, 0};
#if __clang_major__ >= 23
  v8i gz8 = {0, 0, 0, 0, 0, 0, 0, 0};
  __builtin_amdgcn_tensor_load_to_lds(g0, g1, gz, gz, gz8, 0);
#else
  __builtin_amdgcn_tensor_load_to_lds(g0, g1, gz, gz, 0);
#endif
}
#endif

// ---------------------------------------------------------------------------
// Convert W[K][N] f32 -> Wt[N][K] f16 (transposed). grid.y = batch.
// ---------------------------------------------------------------------------
__global__ __launch_bounds__(256) void convT_kernel(
    const float* __restrict__ src, unsigned short* __restrict__ dst,
    int K, int N) {
  const size_t mat = (size_t)K * N;
  const float* s = src + blockIdx.y * mat;
  unsigned short* d = dst + blockIdx.y * mat;
  const int i = blockIdx.x * 256 + threadIdx.x;
  if (i < K * N) {
    const int n = i / K, k = i - n * K;
    d[i] = h16(s[(size_t)k * N + n]);
  }
}

// ---------------------------------------------------------------------------
// Per-row LN statistics over concat [memories[:,l], h]; also emits normalized
// current-token row (qn) and a raw snapshot of h (memory_out).
// ---------------------------------------------------------------------------
__global__ __launch_bounds__(256) void rowstats_kernel(
    const float* __restrict__ mem, const float* __restrict__ h,
    const float* __restrict__ lnS, const float* __restrict__ lnB,
    float* __restrict__ rmean, float* __restrict__ rinv,
    float* __restrict__ qn, float* __restrict__ snap, int layer) {
  const int lane = threadIdx.x & 31, wave = threadIdx.x >> 5;
  const int r = blockIdx.x * 8 + wave;
  const int b = r / S_, s = r - b * S_;
  const float* src = (s < M_)
      ? mem + (((size_t)b * L_ + layer) * M_ + s) * (size_t)E_
      : h + (size_t)b * E_;
  float4 x[4];
  float sum = 0.f, sq = 0.f;
#pragma unroll
  for (int c = 0; c < 4; ++c) {
    x[c] = *(const float4*)(src + lane * 4 + c * 128);
    sum += x[c].x + x[c].y + x[c].z + x[c].w;
    sq  += x[c].x * x[c].x + x[c].y * x[c].y + x[c].z * x[c].z + x[c].w * x[c].w;
  }
#pragma unroll
  for (int off = 16; off; off >>= 1) {
    sum += __shfl_xor(sum, off, 32);
    sq  += __shfl_xor(sq,  off, 32);
  }
  const float mean = sum * (1.0f / E_);
  const float inv  = rsqrtf(sq * (1.0f / E_) - mean * mean + 1e-6f);
  if (lane == 0) { rmean[r] = mean; rinv[r] = inv; }
  if (s == M_) {
#pragma unroll
    for (int c = 0; c < 4; ++c) {
      const int col = lane * 4 + c * 128;
      *(float4*)(snap + (size_t)b * E_ + col) = x[c];  // raw h -> memory_out
      float4 sv = *(const float4*)(lnS + col);
      float4 bv = *(const float4*)(lnB + col);
      float4 y;
      y.x = (x[c].x - mean) * inv * sv.x + bv.x;
      y.y = (x[c].y - mean) * inv * sv.y + bv.y;
      y.z = (x[c].z - mean) * inv * sv.z + bv.z;
      y.w = (x[c].w - mean) * inv * sv.w + bv.w;
      *(float4*)(qn + (size_t)b * E_ + col) = y;
    }
  }
}

// Simple row layernorm (B rows of length E).
__global__ __launch_bounds__(256) void ln_row_kernel(
    const float* __restrict__ src, const float* __restrict__ lnS,
    const float* __restrict__ lnB, float* __restrict__ out) {
  const int lane = threadIdx.x & 31, wave = threadIdx.x >> 5;
  const int r = blockIdx.x * 8 + wave;
  const float* p = src + (size_t)r * E_;
  float4 x[4];
  float sum = 0.f, sq = 0.f;
#pragma unroll
  for (int c = 0; c < 4; ++c) {
    x[c] = *(const float4*)(p + lane * 4 + c * 128);
    sum += x[c].x + x[c].y + x[c].z + x[c].w;
    sq  += x[c].x * x[c].x + x[c].y * x[c].y + x[c].z * x[c].z + x[c].w * x[c].w;
  }
#pragma unroll
  for (int off = 16; off; off >>= 1) {
    sum += __shfl_xor(sum, off, 32);
    sq  += __shfl_xor(sq,  off, 32);
  }
  const float mean = sum * (1.0f / E_);
  const float inv  = rsqrtf(sq * (1.0f / E_) - mean * mean + 1e-6f);
#pragma unroll
  for (int c = 0; c < 4; ++c) {
    const int col = lane * 4 + c * 128;
    float4 sv = *(const float4*)(lnS + col);
    float4 bv = *(const float4*)(lnB + col);
    float4 y;
    y.x = (x[c].x - mean) * inv * sv.x + bv.x;
    y.y = (x[c].y - mean) * inv * sv.y + bv.y;
    y.z = (x[c].z - mean) * inv * sv.z + bv.z;
    y.w = (x[c].w - mean) * inv * sv.w + bv.w;
    *(float4*)(out + (size_t)r * E_ + col) = y;
  }
}

// ---------------------------------------------------------------------------
// Generic GEMM: C = act(A @ W^T16 + bias) (+ residual), W16 is f16 [N][K].
// 64x64 tile, BK=32, 8 waves, each wave a 16x32 strip (2 WMMAs / K-step).
// ---------------------------------------------------------------------------
__global__ __launch_bounds__(256) void gemm_wmma_kernel(
    const float* __restrict__ A, size_t lda,
    const unsigned short* __restrict__ W16, int K, int N, int Mrows,
    const float* __restrict__ bias, const float* __restrict__ res,
    float* __restrict__ C, int relu) {
  __shared__ unsigned short As[64 * 32];
  __shared__ unsigned short Bs[64 * 32];
  unsigned* As32 = (unsigned*)As;
  unsigned* Bs32 = (unsigned*)Bs;

  const int tid = threadIdx.x;
  const int lane = tid & 31, wave = tid >> 5;
  const int wm = wave & 3, wn = wave >> 2;
  const int rowBase = blockIdx.x * 64;
  const int colBase = blockIdx.y * 64;

  const int ar = tid >> 2;        // A tile row 0..63
  const int ac = (tid & 3) << 3;  // A tile k-chunk 0,8,16,24
  const int gr = rowBase + ar;
  const float* asrc = (gr < Mrows) ? A + (size_t)gr * lda : nullptr;

  const int nr = tid >> 2;        // B tile n-row 0..63
  const int kc = (tid & 3) << 3;  // B tile k-chunk
  const int gn = colBase + nr;
  const unsigned* wsrc =
      (gn < N) ? (const unsigned*)(W16 + (size_t)gn * K) : nullptr;

  v8f acc0 = {}, acc1 = {};

  for (int kk = 0; kk < K; kk += 32) {
    float4 f0 = {0, 0, 0, 0}, f1 = {0, 0, 0, 0};
    if (asrc) {
      f0 = *(const float4*)(asrc + kk + ac);
      f1 = *(const float4*)(asrc + kk + ac + 4);
      if (kk + 32 < K) __builtin_prefetch(asrc + kk + 32 + ac, 0, 1);
    }
    {
      unsigned* dst = As32 + ((ar * 32 + ac) >> 1);
      dst[0] = pack2(f0.x, f0.y);
      dst[1] = pack2(f0.z, f0.w);
      dst[2] = pack2(f1.x, f1.y);
      dst[3] = pack2(f1.z, f1.w);
    }
    {
      unsigned z0 = 0, z1 = 0, z2 = 0, z3 = 0;
      if (wsrc) {
        const unsigned* wp = wsrc + ((kk + kc) >> 1);
        z0 = wp[0]; z1 = wp[1]; z2 = wp[2]; z3 = wp[3];
      }
      unsigned* dst = Bs32 + ((nr * 32 + kc) >> 1);
      dst[0] = z0; dst[1] = z1; dst[2] = z2; dst[3] = z3;
    }
    __syncthreads();

    HFrag a, b0, b1;
    {
      const int arow = wm * 16 + (lane & 15);
      const int kadd = (lane & 16) ? 16 : 0;
#pragma unroll
      for (int j = 0; j < 8; ++j) {
        const int koff = 2 * j + kadd;
        a.u[j] = As32[(arow * 32 + koff) >> 1];
      }
      const int n0 = wn * 32 + (lane & 15);
      const int kb = (lane & 16);
#pragma unroll
      for (int j = 0; j < 8; ++j) {
        b0.u[j] = Bs32[(n0 * 32 + kb + 2 * j) >> 1];
        b1.u[j] = Bs32[((n0 + 16) * 32 + kb + 2 * j) >> 1];
      }
    }
    acc0 = __builtin_amdgcn_wmma_f32_16x16x32_f16(false, a.v, false, b0.v,
                                                  (short)0, acc0, false, false);
    acc1 = __builtin_amdgcn_wmma_f32_16x16x32_f16(false, a.v, false, b1.v,
                                                  (short)0, acc1, false, false);
    __syncthreads();
  }

  const int crow = rowBase + wm * 16 + ((lane & 16) ? 8 : 0);
  const int ccol = colBase + wn * 32 + (lane & 15);
#pragma unroll
  for (int r8 = 0; r8 < 8; ++r8) {
    const int row = crow + r8;
    if (row >= Mrows) continue;
    const size_t rb = (size_t)row * N;
    if (ccol < N) {
      float v = acc0[r8] + (bias ? bias[ccol] : 0.f);
      if (relu) v = fmaxf(v, 0.f);
      if (res) v += res[rb + ccol];
      C[rb + ccol] = v;
    }
    if (ccol + 16 < N) {
      float v = acc1[r8] + (bias ? bias[ccol + 16] : 0.f);
      if (relu) v = fmaxf(v, 0.f);
      if (res) v += res[rb + ccol + 16];
      C[rb + ccol + 16] = v;
    }
  }
}

// ---------------------------------------------------------------------------
// Wide K/V projection kernel: each block computes a full 64x512 strip, so the
// big LN(A) operand (135 MB) streams from HBM exactly once per GEMM.
// LayerNorm fused into A staging. W tiles (512x32 f16, pre-transposed [N][K])
// are double-buffered in LDS and fetched by the Tensor Data Mover, with the
// next tile's DMA issued before waiting (s_wait_tensorcnt 1) so the async
// engine overlaps the 16 WMMAs per wave per K-step.
// ---------------------------------------------------------------------------
__global__ __launch_bounds__(256) void kv_wide_kernel(
    const unsigned short* __restrict__ W16,  // [E][E] f16, [n][k]
    int Mrows, float* __restrict__ C,
    const float* __restrict__ mem, const float* __restrict__ hrow,
    const float* __restrict__ rmean, const float* __restrict__ rinv,
    const float* __restrict__ lnS, const float* __restrict__ lnB, int layer) {
  __shared__ unsigned short AsW[64 * 32];       // 4 KB
  __shared__ unsigned short BsW[2][512 * 32];   // 2 x 32 KB (double buffer)
  unsigned* As32 = (unsigned*)AsW;

  const int tid = threadIdx.x;
  const int lane = tid & 31, wave = tid >> 5;
  const int wm = wave & 3, wn = wave >> 2;
  const int rowBase = blockIdx.x * 64;

  const int ar = tid >> 2;
  const int ac = (tid & 3) << 3;
  const int gr = rowBase + ar;
  const int bb = gr / S_, ss = gr - bb * S_;
  const float* asrc = (ss < M_)
      ? mem + (((size_t)bb * L_ + layer) * M_ + ss) * (size_t)E_
      : hrow + (size_t)bb * E_;
  const float amean = rmean[gr];
  const float ainv  = rinv[gr];

  v8f acc[16];
  {
    v8f zero = {};
#pragma unroll
    for (int j = 0; j < 16; ++j) acc[j] = zero;
  }

#if HAVE_TDM
  if (wave == 0) {
    tdm_load_w_tile(W16, (unsigned)(__UINTPTR_TYPE__)&BsW[0][0], E_, E_, E_,
                    E_);
  }
#endif

  for (int kk = 0; kk < E_; kk += 32) {
    const int cur = (kk >> 5) & 1;
    // ---- A tile: load f32, fused layernorm, convert to f16 ----
    float4 f0 = *(const float4*)(asrc + kk + ac);
    float4 f1 = *(const float4*)(asrc + kk + ac + 4);
    if (kk + 32 < E_) __builtin_prefetch(asrc + kk + 32 + ac, 0, 1);
    float4 s0 = *(const float4*)(lnS + kk + ac);
    float4 s1 = *(const float4*)(lnS + kk + ac + 4);
    float4 b0 = *(const float4*)(lnB + kk + ac);
    float4 b1 = *(const float4*)(lnB + kk + ac + 4);
    f0.x = (f0.x - amean) * ainv * s0.x + b0.x;
    f0.y = (f0.y - amean) * ainv * s0.y + b0.y;
    f0.z = (f0.z - amean) * ainv * s0.z + b0.z;
    f0.w = (f0.w - amean) * ainv * s0.w + b0.w;
    f1.x = (f1.x - amean) * ainv * s1.x + b1.x;
    f1.y = (f1.y - amean) * ainv * s1.y + b1.y;
    f1.z = (f1.z - amean) * ainv * s1.z + b1.z;
    f1.w = (f1.w - amean) * ainv * s1.w + b1.w;
    {
      unsigned* dst = As32 + ((ar * 32 + ac) >> 1);
      dst[0] = pack2(f0.x, f0.y);
      dst[1] = pack2(f0.z, f0.w);
      dst[2] = pack2(f1.x, f1.y);
      dst[3] = pack2(f1.z, f1.w);
    }
    // ---- B tile: pipelined Tensor DMA (or manual fallback) ----
#if HAVE_TDM
    if (wave == 0) {
      if (kk + 32 < E_) {
        tdm_load_w_tile(W16 + kk + 32,
                        (unsigned)(__UINTPTR_TYPE__)&BsW[cur ^ 1][0], E_,
                        E_ - (kk + 32), E_, E_);
        __builtin_amdgcn_s_wait_tensorcnt(1);  // current tile complete
      } else {
        __builtin_amdgcn_s_wait_tensorcnt(0);
      }
    }
#else
    {
      unsigned* bdst = (unsigned*)&BsW[cur][0];
#pragma unroll
      for (int r = 0; r < 2; ++r) {
        const int n = tid * 2 + r;
        const unsigned* wp = (const unsigned*)(W16 + (size_t)n * E_ + kk);
        unsigned* dst = bdst + n * 16;
#pragma unroll
        for (int i = 0; i < 16; ++i) dst[i] = wp[i];
      }
    }
#endif
    __syncthreads();

    HFrag a;
    {
      const int arow = wm * 16 + (lane & 15);
      const int kadd = (lane & 16) ? 16 : 0;
#pragma unroll
      for (int j = 0; j < 8; ++j) {
        const int koff = 2 * j + kadd;
        a.u[j] = As32[(arow * 32 + koff) >> 1];
      }
    }
    const unsigned* Bs32 = (const unsigned*)&BsW[cur][0];
    const int kb = (lane & 16);
    const int nlane = (lane & 15);
#pragma unroll
    for (int j = 0; j < 16; ++j) {
      HFrag b;
      const int n0 = wn * 256 + j * 16 + nlane;
#pragma unroll
      for (int i = 0; i < 8; ++i)
        b.u[i] = Bs32[(n0 * 32 + kb + 2 * i) >> 1];
      acc[j] = __builtin_amdgcn_wmma_f32_16x16x32_f16(
          false, a.v, false, b.v, (short)0, acc[j], false, false);
    }
    __syncthreads();
  }

  const int crow = rowBase + wm * 16 + ((lane & 16) ? 8 : 0);
  const int nlane = (lane & 15);
#pragma unroll
  for (int j = 0; j < 16; ++j) {
    const int col = wn * 256 + j * 16 + nlane;
#pragma unroll
    for (int r8 = 0; r8 < 8; ++r8) {
      const int row = crow + r8;
      if (row < Mrows) C[(size_t)row * E_ + col] = acc[j][r8];
    }
  }
}

// ---------------------------------------------------------------------------
// Attention: one wave per (b, head). Online softmax, float2/lane over DH=64.
// ---------------------------------------------------------------------------
__global__ __launch_bounds__(256) void attn_kernel(
    const float* __restrict__ Q, const float* __restrict__ Kp,
    const float* __restrict__ Vp, const unsigned char* __restrict__ mask,
    float* __restrict__ O) {
  const int lane = threadIdx.x & 31, wave = threadIdx.x >> 5;
  const int idx = blockIdx.x * 8 + wave;
  const int b = idx >> 3, hh = idx & 7;
  const float2 q2 = *(const float2*)(Q + (size_t)b * E_ + hh * DH_ + lane * 2);
  const float* kbase = Kp + (size_t)b * S_ * E_ + hh * DH_ + lane * 2;
  const float* vbase = Vp + (size_t)b * S_ * E_ + hh * DH_ + lane * 2;
  float mx = -1e30f, sum = 0.f, ox = 0.f, oy = 0.f;
  for (int s = 0; s < S_; ++s) {
    const float2 k2 = *(const float2*)(kbase + (size_t)s * E_);
    float p = q2.x * k2.x + q2.y * k2.y;
#pragma unroll
    for (int off = 16; off; off >>= 1) p += __shfl_xor(p, off, 32);
    p *= 0.125f;  // 1/sqrt(DH)
    if (s < M_ && !mask[(size_t)b * M_ + s]) p = -1e9f;
    const float2 v2 = *(const float2*)(vbase + (size_t)s * E_);
    if (p > mx) {
      const float c = __expf(mx - p);
      sum *= c; ox *= c; oy *= c; mx = p;
    }
    const float e = __expf(p - mx);
    sum += e; ox += e * v2.x; oy += e * v2.y;
  }
  const float r = 1.0f / sum;
  float2 o2; o2.x = ox * r; o2.y = oy * r;
  *(float2*)(O + (size_t)b * E_ + hh * DH_ + lane * 2) = o2;
}

__global__ __launch_bounds__(256) void gather_kernel(
    const float* __restrict__ hbuf, const float* __restrict__ hsnap,
    const int* __restrict__ skills, float* __restrict__ xsel,
    float* __restrict__ memout) {
  const int i = blockIdx.x * 256 + threadIdx.x;  // over B*L*E
  const int b = i / (L_ * E_);
  const int rem = i - b * (L_ * E_);
  const int l = rem / E_, e = rem - l * E_;
  const int tt = skills[b];
  memout[i] = hsnap[(((size_t)tt * L_ + l) * B_ + b) * E_ + e];
  if (i < B_ * E_) {
    const int b2 = i / E_, e2 = i - b2 * E_;
    const int t2 = skills[b2];
    xsel[i] = hbuf[((size_t)t2 * B_ + b2) * E_ + e2];
  }
}

__global__ __launch_bounds__(256) void final_select_kernel(
    const float* __restrict__ abuf, const float* __restrict__ cbuf,
    const int* __restrict__ skills, float* __restrict__ logits,
    float* __restrict__ vals) {
  const int i = blockIdx.x * 256 + threadIdx.x;  // over B*A
  const int b = i / A_, a = i - b * A_;
  const int tt = skills[b];
  logits[i] = abuf[((size_t)tt * B_ + b) * A_ + a];
  if (i < B_) vals[i] = cbuf[(size_t)skills[i] * B_ + i];
}

// ---------------------------------------------------------------------------
extern "C" void kernel_launch(void* const* d_in, const int* in_sizes, int n_in,
                              void* d_out, int out_size, void* d_ws,
                              size_t ws_size, hipStream_t stream) {
  (void)in_sizes; (void)n_in; (void)out_size; (void)ws_size;
  const float* memories = (const float*)d_in[0];
  const float* obs  = (const float*)d_in[1];
  const float* We   = (const float*)d_in[2];
  const float* be   = (const float*)d_in[3];
  const float* Wq   = (const float*)d_in[4];
  const float* Wk   = (const float*)d_in[5];
  const float* Wv   = (const float*)d_in[6];
  const float* Wo   = (const float*)d_in[7];
  const float* bo   = (const float*)d_in[8];
  const float* W1   = (const float*)d_in[9];
  const float* b1   = (const float*)d_in[10];
  const float* W2   = (const float*)d_in[11];
  const float* b2   = (const float*)d_in[12];
  const float* ln1s = (const float*)d_in[13];
  const float* ln1b = (const float*)d_in[14];
  const float* ln2s = (const float*)d_in[15];
  const float* ln2b = (const float*)d_in[16];
  const float* aW1  = (const float*)d_in[17];
  const float* ab1  = (const float*)d_in[18];
  const float* aW2  = (const float*)d_in[19];
  const float* ab2  = (const float*)d_in[20];
  const float* aW3  = (const float*)d_in[21];
  const float* ab3  = (const float*)d_in[22];
  const float* cW1  = (const float*)d_in[23];
  const float* cb1  = (const float*)d_in[24];
  const float* cW2  = (const float*)d_in[25];
  const float* cb2  = (const float*)d_in[26];
  const float* cW3  = (const float*)d_in[27];
  const float* cb3  = (const float*)d_in[28];
  const int* skills = (const int*)d_in[29];
  const unsigned char* mask = (const unsigned char*)d_in[30];

  // ---- workspace layout: f32 scratch then f16 transposed weights (~410MB)
  float* ws = (float*)d_ws;
  float* RM    = ws; ws += (size_t)B_ * S_;
  float* RI    = ws; ws += (size_t)B_ * S_;
  float* QN    = ws; ws += (size_t)B_ * E_;
  float* KP    = ws; ws += (size_t)B_ * S_ * E_;
  float* VP    = ws; ws += (size_t)B_ * S_ * E_;
  float* QP    = ws; ws += (size_t)B_ * E_;
  float* AO    = ws; ws += (size_t)B_ * E_;
  float* HN    = ws; ws += (size_t)B_ * E_;
  float* MLP1  = ws; ws += (size_t)B_ * E_;
  float* XSEL  = ws; ws += (size_t)B_ * E_;
  float* HBUF  = ws; ws += (size_t)T_ * B_ * E_;
  float* HSNAP = ws; ws += (size_t)T_ * L_ * B_ * E_;
  float* AH1   = ws; ws += (size_t)B_ * HID_;
  float* AH2   = ws; ws += (size_t)B_ * HID_;
  float* ABUF  = ws; ws += (size_t)T_ * B_ * A_;
  float* CBUF  = ws; ws += (size_t)T_ * B_;
  unsigned short* wh = (unsigned short*)ws;
  unsigned short* WeT  = wh; wh += (size_t)T_ * E_ * OBS_;
  unsigned short* WqT  = wh; wh += (size_t)T_ * L_ * E_ * E_;
  unsigned short* WkT  = wh; wh += (size_t)T_ * L_ * E_ * E_;
  unsigned short* WvT  = wh; wh += (size_t)T_ * L_ * E_ * E_;
  unsigned short* WoT  = wh; wh += (size_t)T_ * L_ * E_ * E_;
  unsigned short* W1T  = wh; wh += (size_t)T_ * L_ * E_ * E_;
  unsigned short* W2T  = wh; wh += (size_t)T_ * L_ * E_ * E_;
  unsigned short* aW1T = wh; wh += (size_t)T_ * HID_ * E_;
  unsigned short* aW2T = wh; wh += (size_t)T_ * HID_ * HID_;
  unsigned short* aW3T = wh; wh += (size_t)T_ * A_ * HID_;
  unsigned short* cW1T = wh; wh += (size_t)T_ * HID_ * E_;
  unsigned short* cW2T = wh; wh += (size_t)T_ * HID_ * HID_;
  unsigned short* cW3T = wh; wh += (size_t)T_ * HID_;

  const dim3 blk(256, 1, 1);
  auto convT = [&](const float* src, unsigned short* dst, int K, int N,
                   int batch) {
    dim3 g((K * N + 255) / 256, batch, 1);
    convT_kernel<<<g, blk, 0, stream>>>(src, dst, K, N);
  };
  convT(We, WeT, OBS_, E_, T_);
  convT(Wq, WqT, E_, E_, T_ * L_);
  convT(Wk, WkT, E_, E_, T_ * L_);
  convT(Wv, WvT, E_, E_, T_ * L_);
  convT(Wo, WoT, E_, E_, T_ * L_);
  convT(W1, W1T, E_, E_, T_ * L_);
  convT(W2, W2T, E_, E_, T_ * L_);
  convT(aW1, aW1T, E_, HID_, T_);
  convT(aW2, aW2T, HID_, HID_, T_);
  convT(aW3, aW3T, HID_, A_, T_);
  convT(cW1, cW1T, E_, HID_, T_);
  convT(cW2, cW2T, HID_, HID_, T_);
  convT(cW3, cW3T, HID_, 1, T_);

  auto gemm = [&](const float* Ap, size_t lda, const unsigned short* Wp, int K,
                  int N, int Mr, const float* bias, const float* res,
                  float* Cp, int relu) {
    dim3 g((Mr + 63) / 64, (N + 63) / 64, 1);
    gemm_wmma_kernel<<<g, blk, 0, stream>>>(Ap, lda, Wp, K, N, Mr, bias, res,
                                            Cp, relu);
  };

  for (int t = 0; t < T_; ++t) {
    float* h = HBUF + (size_t)t * B_ * E_;
    gemm(obs, OBS_, WeT + (size_t)t * E_ * OBS_, OBS_, E_, B_, be + t * E_,
         nullptr, h, 0);
    for (int l = 0; l < L_; ++l) {
      const size_t tl = (size_t)t * L_ + l;
      const float* lS1 = ln1s + tl * E_;
      const float* lB1 = ln1b + tl * E_;
      float* snap = HSNAP + tl * B_ * E_;
      rowstats_kernel<<<dim3(B_ * S_ / 8), blk, 0, stream>>>(
          memories, h, lS1, lB1, RM, RI, QN, snap, l);
      gemm(QN, E_, WqT + tl * E_ * E_, E_, E_, B_, nullptr, nullptr, QP, 0);
      kv_wide_kernel<<<dim3(B_ * S_ / 64), blk, 0, stream>>>(
          WkT + tl * E_ * E_, B_ * S_, KP, memories, h, RM, RI, lS1, lB1, l);
      kv_wide_kernel<<<dim3(B_ * S_ / 64), blk, 0, stream>>>(
          WvT + tl * E_ * E_, B_ * S_, VP, memories, h, RM, RI, lS1, lB1, l);
      attn_kernel<<<dim3(B_ * H_ / 8), blk, 0, stream>>>(QP, KP, VP, mask, AO);
      gemm(AO, E_, WoT + tl * E_ * E_, E_, E_, B_, bo + tl * E_, h, h, 0);
      ln_row_kernel<<<dim3(B_ / 8), blk, 0, stream>>>(h, ln2s + tl * E_,
                                                      ln2b + tl * E_, HN);
      gemm(HN, E_, W1T + tl * E_ * E_, E_, E_, B_, b1 + tl * E_, nullptr,
           MLP1, 1);
      gemm(MLP1, E_, W2T + tl * E_ * E_, E_, E_, B_, b2 + tl * E_, h, h, 0);
    }
  }

  float* out_logits = (float*)d_out;
  float* out_vals   = out_logits + (size_t)B_ * A_;
  float* out_mem    = out_vals + B_;
  gather_kernel<<<dim3(B_ * L_ * E_ / 256), blk, 0, stream>>>(
      HBUF, HSNAP, skills, XSEL, out_mem);

  for (int t = 0; t < T_; ++t) {
    gemm(XSEL, E_, aW1T + (size_t)t * HID_ * E_, E_, HID_, B_, ab1 + t * HID_,
         nullptr, AH1, 1);
    gemm(AH1, HID_, aW2T + (size_t)t * HID_ * HID_, HID_, HID_, B_,
         ab2 + t * HID_, nullptr, AH2, 1);
    gemm(AH2, HID_, aW3T + (size_t)t * A_ * HID_, HID_, A_, B_, ab3 + t * A_,
         nullptr, ABUF + (size_t)t * B_ * A_, 0);
    gemm(XSEL, E_, cW1T + (size_t)t * HID_ * E_, E_, HID_, B_, cb1 + t * HID_,
         nullptr, AH1, 1);
    gemm(AH1, HID_, cW2T + (size_t)t * HID_ * HID_, HID_, HID_, B_,
         cb2 + t * HID_, nullptr, AH2, 1);
    gemm(AH2, HID_, cW3T + (size_t)t * HID_, HID_, 1, B_, cb3 + t, nullptr,
         CBUF + (size_t)t * B_, 0);
  }

  final_select_kernel<<<dim3(B_ * A_ / 256), blk, 0, stream>>>(
      ABUF, CBUF, skills, out_logits, out_vals);
}
